// SelfAttention_30571577213139
// MI455X (gfx1250) — matrix operand order repaired and checked by reading, not compiled, not run on hardware
//
#include <hip/hip_runtime.h>

// ---------------------------------------------------------------------------
// Self-attention for MI455X (gfx1250, wave32, WMMA, TDM).
//   1) cast x, qkv_w, out_w -> bf16
//   2) qkv = x @ qkv_w^T + qkv_b   (WMMA bf16 GEMM; B tile staged to LDS by
//      the Tensor Data Mover, double-buffered with s_wait_tensorcnt)
//   3) RoPE on q,k; pack q,k -> [B,H,T,64], v -> transposed [B,H,64,T]
//   4) flash attention per (b,h,16-row q tile): WMMA S=QK^T and O=PV,
//      online softmax via wave32 shfl reductions, LDS S->P relayout
//   5) out = ctx @ out_w^T + out_b (same TDM GEMM)
// ---------------------------------------------------------------------------

#define HIDDEN 1024
#define HEADS  16
#define HDIM   64
#define BB     2
#define TT     2048
#define M_ROWS (BB * TT)          // 4096
#define NEG_INF (-3.0e38f)

typedef __bf16 bf16_t;
typedef bf16_t v8bf  __attribute__((ext_vector_type(8)));
typedef bf16_t v16bf __attribute__((ext_vector_type(16)));
typedef float  v8f   __attribute__((ext_vector_type(8)));
typedef unsigned int u32x4 __attribute__((ext_vector_type(4)));
typedef int          i32x4 __attribute__((ext_vector_type(4)));
typedef int          i32x8 __attribute__((ext_vector_type(8)));

#if defined(__HIP_DEVICE_COMPILE__) && __has_builtin(__builtin_amdgcn_tensor_load_to_lds)
#define HAVE_TDM 1
#else
#define HAVE_TDM 0
#endif

// Build a 16-element bf16 fragment from two contiguous 16B chunks.
__device__ __forceinline__ v16bf frag16(const bf16_t* p0, const bf16_t* p1) {
  v8bf a = *(const v8bf*)p0;
  v8bf b = *(const v8bf*)p1;
  return __builtin_shufflevector(a, b, 0,1,2,3,4,5,6,7,8,9,10,11,12,13,14,15);
}

__device__ __forceinline__ v8f wmma_bf16(v16bf a, v16bf b, v8f c) {
  return __builtin_amdgcn_wmma_f32_16x16x32_bf16(false, a, false, b,
                                                 (short)0, c, false, false);
}

// TDM: DMA a 2D bf16 tile (tile_x elems along dim0 x tile_y rows, row stride
// stride_elems) from global into LDS at byte offset lds_off (rows compacted).
// D# packing per CDNA5 ISA ch.8 (group0: count/lds/global/type, group1: dims).
__device__ __forceinline__ void tdm_load_2d(unsigned lds_off, const void* gaddr,
                                            int tile_x, int tile_y,
                                            int stride_elems) {
#if HAVE_TDM
  unsigned long long ga = (unsigned long long)(size_t)gaddr;
  u32x4 g0;
  g0[0] = 0x1u;                                      // count=1, user mode
  g0[1] = lds_off;                                   // lds_addr (bytes)
  g0[2] = (unsigned)(ga & 0xFFFFFFFFu);              // global_addr[31:0]
  g0[3] = (unsigned)((ga >> 32) & 0x01FFFFFFu) | (2u << 30);  // addr[56:32]|type=2
  i32x8 g1;
  g1[0] = (1 << 16);                                 // data_size=1 (2B), wg_mask=0
  g1[1] = (tile_x & 0xFFFF) << 16;                   // tensor_dim0[15:0]
  g1[2] = ((tile_x >> 16) & 0xFFFF) | ((tile_y & 0xFFFF) << 16); // d0 hi | d1 lo
  g1[3] = ((tile_y >> 16) & 0xFFFF) | ((tile_x & 0xFFFF) << 16); // d1 hi | tile_dim0
  g1[4] = (tile_y & 0xFFFF);                         // tile_dim1, tile_dim2=0
  g1[5] = stride_elems;                              // tensor_dim0_stride[31:0]
  g1[6] = 0;
  g1[7] = 0;
  i32x4 z4 = {0, 0, 0, 0};
#if __clang_major__ >= 23
  i32x8 z8 = {0, 0, 0, 0, 0, 0, 0, 0};
  __builtin_amdgcn_tensor_load_to_lds(g0, g1, z4, z4, z8, 0);
#else
  __builtin_amdgcn_tensor_load_to_lds(g0, g1, z4, z4, 0);
#endif
#endif
}

__device__ __forceinline__ void wait_tensorcnt_le(int n) {
#if defined(__HIP_DEVICE_COMPILE__) && __has_builtin(__builtin_amdgcn_s_wait_tensorcnt)
  if (n) __builtin_amdgcn_s_wait_tensorcnt(1);
  else   __builtin_amdgcn_s_wait_tensorcnt(0);
#endif
}

// ---------------------------------------------------------------------------
__global__ void cast_f32_bf16(const float* __restrict__ src,
                              bf16_t* __restrict__ dst, int n) {
  int i = blockIdx.x * blockDim.x + threadIdx.x;
  if (i < n) dst[i] = (bf16_t)src[i];
}

// ---------------------------------------------------------------------------
// C[M,N] = A[M,K] @ Wt[N,K]^T + bias[N].
// Block = 8 waves = 128x64 tile; shared 64x32 B tile staged into LDS by TDM
// (wave 0), double-buffered; each wave owns a 16x64 slice (1 A-frag, 4 WMMA
// per K-step).
__global__ __launch_bounds__(256) void gemm_bf16_nt(
    const bf16_t* __restrict__ A, const bf16_t* __restrict__ Wt,
    const float* __restrict__ bias, float* __restrict__ C,
    int M, int N, int K) {
  __shared__ bf16_t sB[2][64 * 32];                  // 2 x 4 KB

  int ntn = N >> 6;
  int tn  = blockIdx.x % ntn;
  int wave = threadIdx.x >> 5;
  int tm  = (int)(blockIdx.x / ntn) * 8 + wave;      // 16-row tile index
  int l = threadIdx.x & 31;
  int lm = l & 15, half = l >> 4;

  const bf16_t* arow  = A + (size_t)(tm * 16 + lm) * K;
  const bf16_t* btile = Wt + (size_t)(tn * 64) * K;

  v8f acc0 = {}, acc1 = {}, acc2 = {}, acc3 = {};

#if HAVE_TDM
  if (wave == 0)
    tdm_load_2d((unsigned)(size_t)&sB[0][0], btile, 32, 64, K);
#endif

  int ib = 0;
  for (int kk = 0; kk < K; kk += 32) {
#if HAVE_TDM
    if (wave == 0) {
      if (kk + 32 < K) {
        tdm_load_2d((unsigned)(size_t)&sB[ib ^ 1][0], btile + kk + 32, 32, 64, K);
        wait_tensorcnt_le(1);      // in-order TDM: current buffer is complete
      } else {
        wait_tensorcnt_le(0);
      }
    }
#else
    {  // cooperative fallback staging: 256 threads x 16B = 4 KB
      int t = threadIdx.x;
      *(v8bf*)&sB[ib][(t >> 2) * 32 + (t & 3) * 8] =
          *(const v8bf*)(btile + (size_t)(t >> 2) * K + kk + (t & 3) * 8);
    }
#endif
    __syncthreads();               // B tile visible to all 8 waves

    if (kk + 32 < K) __builtin_prefetch(arow + kk + 32, 0, 1);
    v16bf a = frag16(arow + kk + 8 * half, arow + kk + 16 + 8 * half);
    const bf16_t* bp = &sB[ib][(size_t)lm * 32 + 8 * half];
    acc0 = wmma_bf16(a, frag16(bp,           bp + 16),           acc0);
    acc1 = wmma_bf16(a, frag16(bp + 16 * 32, bp + 16 * 32 + 16), acc1);
    acc2 = wmma_bf16(a, frag16(bp + 32 * 32, bp + 32 * 32 + 16), acc2);
    acc3 = wmma_bf16(a, frag16(bp + 48 * 32, bp + 48 * 32 + 16), acc3);

    __syncthreads();               // all reads done before buffer is reused
    ib ^= 1;
  }

  int cb = tn * 64 + lm;
#pragma unroll
  for (int r = 0; r < 8; ++r) {
    float* cr = C + (size_t)(tm * 16 + r + 8 * half) * N;
    cr[cb]      = acc0[r] + bias[cb];
    cr[cb + 16] = acc1[r] + bias[cb + 16];
    cr[cb + 32] = acc2[r] + bias[cb + 32];
    cr[cb + 48] = acc3[r] + bias[cb + 48];
  }
}

// ---------------------------------------------------------------------------
// RoPE + repack: qkv[M,3072] -> qb/kb [B,H,T,64], vt [B,H,64,T]; global mask.
__global__ void rope_pack_kernel(const float* __restrict__ qkv,
                                 const int* __restrict__ ids,
                                 const long long* __restrict__ gidx, int n_g,
                                 bf16_t* __restrict__ qb, bf16_t* __restrict__ kb,
                                 bf16_t* __restrict__ vt,
                                 unsigned char* __restrict__ gmask) {
  int tid = blockIdx.x * blockDim.x + threadIdx.x;
  if (tid >= BB * TT * HEADS * 32) return;
  int i = tid & 31;
  int h = (tid >> 5) & (HEADS - 1);
  int t = (tid >> 9) & (TT - 1);
  int b = tid >> 20;
  int row = b * TT + t;
  const float* r = qkv + (size_t)row * (3 * HIDDEN);

  float inv = __powf(10000.0f, -(float)i * (1.0f / 32.0f));
  float s, c;
  __sincosf((float)t * inv, &s, &c);

  float q1 = r[h * 64 + i],              q2 = r[h * 64 + 32 + i];
  float k1 = r[HIDDEN + h * 64 + i],     k2 = r[HIDDEN + h * 64 + 32 + i];
  float v1 = r[2 * HIDDEN + h * 64 + i], v2 = r[2 * HIDDEN + h * 64 + 32 + i];

  size_t ho = ((size_t)(b * HEADS + h) * TT + t) * HDIM;
  qb[ho + i]      = (bf16_t)(q1 * c - q2 * s);
  qb[ho + 32 + i] = (bf16_t)(q1 * s + q2 * c);
  kb[ho + i]      = (bf16_t)(k1 * c - k2 * s);
  kb[ho + 32 + i] = (bf16_t)(k1 * s + k2 * c);

  size_t vo = (size_t)(b * HEADS + h) * HDIM * TT + t;
  vt[vo + (size_t)i * TT]        = (bf16_t)v1;
  vt[vo + (size_t)(32 + i) * TT] = (bf16_t)v2;

  if (h == 0 && i == 0) {
    int id = ids[row];
    unsigned char g = 0;
    for (int j = 0; j < n_g; ++j) g |= (id == (int)gidx[j]);
    gmask[row] = g;
  }
}

// ---------------------------------------------------------------------------
// Flash attention: one wave (blockDim=32) per (b,h,16-row q tile).
__global__ __launch_bounds__(32) void attn_kernel(
    const bf16_t* __restrict__ qb, const bf16_t* __restrict__ kmat,
    const bf16_t* __restrict__ vt, const unsigned char* __restrict__ gmask,
    bf16_t* __restrict__ ctx) {
  __shared__ bf16_t sP[16][32];     // S->P fragment relayout buffer (1 KB)

  const int ntq = TT / 16;
  int qt = blockIdx.x % ntq;
  int bh = blockIdx.x / ntq;
  int b = bh / HEADS, h = bh % HEADS;
  int l = threadIdx.x, lm = l & 15, half = l >> 4;
  int qbase = qt * 16;

  const bf16_t* qrow = qb + ((size_t)bh * TT + qbase + lm) * HDIM;
  v16bf qa0 = frag16(qrow + 8 * half,      qrow + 16 + 8 * half);
  v16bf qa1 = frag16(qrow + 32 + 8 * half, qrow + 48 + 8 * half);

  unsigned long long gfl =
      *(const unsigned long long*)(gmask + b * TT + qbase + 8 * half);
  bool anyg = (__ballot(gfl != 0ULL) != 0ULL);           // wave-uniform
  int kEnd = anyg ? TT : ((qbase + 16 + 31) & ~31);      // causal early-exit

  float mr[8], lr[8];
  v8f o0 = {}, o1 = {}, o2 = {}, o3 = {};
#pragma unroll
  for (int r = 0; r < 8; ++r) { mr[r] = NEG_INF; lr[r] = 0.0f; }

  const float sc = 0.125f;          // 1/sqrt(64)
  for (int k0 = 0; k0 < kEnd; k0 += 32) {
    const bf16_t* kr0 = kmat + ((size_t)bh * TT + k0 + lm) * HDIM + 8 * half;
    const bf16_t* kr1 = kr0 + 16 * HDIM;
    v8f s0 = {}, s1 = {};
    s0 = wmma_bf16(qa0, frag16(kr0,      kr0 + 16),  s0);
    s0 = wmma_bf16(qa1, frag16(kr0 + 32, kr0 + 48), s0);
    s1 = wmma_bf16(qa0, frag16(kr1,      kr1 + 16),  s1);
    s1 = wmma_bf16(qa1, frag16(kr1 + 32, kr1 + 48), s1);

    int c0 = k0 + lm, c1 = k0 + 16 + lm;
#pragma unroll
    for (int r = 0; r < 8; ++r) {
      int qg = qbase + r + 8 * half;
      bool g = ((gfl >> (8 * r)) & 1ULL) != 0ULL;
      float a0 = (g || c0 <= qg) ? s0[r] * sc : NEG_INF;
      float a1 = (g || c1 <= qg) ? s1[r] * sc : NEG_INF;
      float v = fmaxf(a0, a1);
      v = fmaxf(v, __shfl_xor(v, 1, 32));
      v = fmaxf(v, __shfl_xor(v, 2, 32));
      v = fmaxf(v, __shfl_xor(v, 4, 32));
      v = fmaxf(v, __shfl_xor(v, 8, 32));
      float mnew = fmaxf(mr[r], v);
      float p0 = __expf(a0 - mnew), p1 = __expf(a1 - mnew);
      float ps = p0 + p1;
      ps += __shfl_xor(ps, 1, 32);
      ps += __shfl_xor(ps, 2, 32);
      ps += __shfl_xor(ps, 4, 32);
      ps += __shfl_xor(ps, 8, 32);
      float corr = __expf(mr[r] - mnew);
      lr[r] = lr[r] * corr + ps;
      mr[r] = mnew;
      o0[r] *= corr; o1[r] *= corr; o2[r] *= corr; o3[r] *= corr;
      int prow = r + 8 * half;
      sP[prow][lm]      = (bf16_t)p0;
      sP[prow][lm + 16] = (bf16_t)p1;
    }
    __syncthreads();                // single-wave WG: DS in-order, barrier ~NOP
    v16bf pf = frag16(&sP[lm][8 * half], &sP[lm][16 + 8 * half]);
    __syncthreads();

    const bf16_t* vr = vt + ((size_t)bh * HDIM + lm) * TT + k0 + 8 * half;
    o0 = wmma_bf16(pf, frag16(vr, vr + 16), o0); vr += (size_t)16 * TT;
    o1 = wmma_bf16(pf, frag16(vr, vr + 16), o1); vr += (size_t)16 * TT;
    o2 = wmma_bf16(pf, frag16(vr, vr + 16), o2); vr += (size_t)16 * TT;
    o3 = wmma_bf16(pf, frag16(vr, vr + 16), o3);
  }

#pragma unroll
  for (int r = 0; r < 8; ++r) {
    float inv = 1.0f / lr[r];
    size_t row = (size_t)b * TT + qbase + r + 8 * half;
    bf16_t* cr = ctx + row * HIDDEN + h * HDIM + lm;
    cr[0]  = (bf16_t)(o0[r] * inv);
    cr[16] = (bf16_t)(o1[r] * inv);
    cr[32] = (bf16_t)(o2[r] * inv);
    cr[48] = (bf16_t)(o3[r] * inv);
  }
}

// ---------------------------------------------------------------------------
extern "C" void kernel_launch(void* const* d_in, const int* in_sizes, int n_in,
                              void* d_out, int out_size, void* d_ws, size_t ws_size,
                              hipStream_t stream) {
  const float*     x     = (const float*)d_in[0];
  const int*       ids   = (const int*)d_in[1];
  const float*     qkv_w = (const float*)d_in[2];
  const float*     qkv_b = (const float*)d_in[3];
  const float*     out_w = (const float*)d_in[4];
  const float*     out_b = (const float*)d_in[5];
  const long long* gidx  = (const long long*)d_in[6];
  int n_g = in_sizes[6];
  float* out = (float*)d_out;

  char* p = (char*)d_ws;
  auto alloc = [&](size_t bytes) {
    char* r = p;
    p += (bytes + 255) & ~(size_t)255;
    return r;
  };
  bf16_t* xb   = (bf16_t*)alloc((size_t)M_ROWS * HIDDEN * 2);
  bf16_t* wq   = (bf16_t*)alloc((size_t)3 * HIDDEN * HIDDEN * 2);
  bf16_t* wo   = (bf16_t*)alloc((size_t)HIDDEN * HIDDEN * 2);
  float*  qkvf = (float*) alloc((size_t)M_ROWS * 3 * HIDDEN * 4);
  bf16_t* qb   = (bf16_t*)alloc((size_t)M_ROWS * HIDDEN * 2);
  bf16_t* kbuf = (bf16_t*)alloc((size_t)M_ROWS * HIDDEN * 2);
  bf16_t* vt   = (bf16_t*)alloc((size_t)M_ROWS * HIDDEN * 2);
  bf16_t* ctx  = (bf16_t*)alloc((size_t)M_ROWS * HIDDEN * 2);
  unsigned char* gm = (unsigned char*)alloc(M_ROWS);

  int n1 = M_ROWS * HIDDEN;
  cast_f32_bf16<<<(n1 + 255) / 256, 256, 0, stream>>>(x, xb, n1);
  int n2 = 3 * HIDDEN * HIDDEN;
  cast_f32_bf16<<<(n2 + 255) / 256, 256, 0, stream>>>(qkv_w, wq, n2);
  int n3 = HIDDEN * HIDDEN;
  cast_f32_bf16<<<(n3 + 255) / 256, 256, 0, stream>>>(out_w, wo, n3);

  // qkv = x @ qkv_w^T + qkv_b : blocks = (4096/128)*(3072/64) = 1536
  gemm_bf16_nt<<<(M_ROWS / 128) * (3 * HIDDEN / 64), 256, 0, stream>>>(
      xb, wq, qkv_b, qkvf, M_ROWS, 3 * HIDDEN, HIDDEN);

  rope_pack_kernel<<<(BB * TT * HEADS * 32) / 256, 256, 0, stream>>>(
      qkvf, ids, gidx, n_g, qb, kbuf, vt, gm);

  attn_kernel<<<BB * HEADS * (TT / 16), 32, 0, stream>>>(qb, kbuf, vt, gm, ctx);

  // out = ctx @ out_w^T + out_b : blocks = (4096/128)*(1024/64) = 512
  gemm_bf16_nt<<<(M_ROWS / 128) * (HIDDEN / 64), 256, 0, stream>>>(
      ctx, wo, out_b, out, M_ROWS, HIDDEN, HIDDEN);
}
